// LightNetAttention_13752485282229
// MI455X (gfx1250) — matrix operand order repaired and checked by reading, not compiled, not run on hardware
//
#include <hip/hip_runtime.h>

typedef __attribute__((ext_vector_type(16))) __bf16 v16bf;
typedef __attribute__((ext_vector_type(8)))  float  v8f;
typedef int i32x4 __attribute__((vector_size(16)));   // matches async builtin proto

#define B_    2
#define T_    2048
#define HID_  1024
#define H_    8
#define DK_   128
#define DV_   128
#define LR_   128
#define CHK_  64
#define NC_   (T_ / CHK_)
#define BH_   (B_ * H_)
#define M_    (B_ * T_)
#define EPS_  1e-5f
#define SCALE_ 0.08838834764831845f   // DK^-0.5

#ifndef __has_builtin
#define __has_builtin(x) 0
#endif
#if __has_builtin(__builtin_amdgcn_global_load_async_to_lds_b128)
#define HAVE_ASYNC 1
#else
#define HAVE_ASYNC 0
#endif

#if HAVE_ASYNC
#if __has_builtin(__builtin_amdgcn_s_wait_asynccnt)
#define WAIT_ASYNC(n) __builtin_amdgcn_s_wait_asynccnt(n)
#else
#define WAIT_ASYNC(n) asm volatile("s_wait_asynccnt " #n ::: "memory")
#endif
#else
#define WAIT_ASYNC(n)
#endif

#if __has_builtin(__builtin_amdgcn_tensor_load_to_lds) && \
    __has_builtin(__builtin_amdgcn_s_wait_tensorcnt)
#define HAVE_TDM 1
#else
#define HAVE_TDM 0
#endif

// 16-byte global -> LDS copy: async DMA on CDNA5, plain vector copy otherwise
__device__ __forceinline__ void cp16(const __bf16* g, __bf16* l) {
#if HAVE_ASYNC
  __builtin_amdgcn_global_load_async_to_lds_b128(
      (__attribute__((address_space(1))) i32x4*)(g),
      (__attribute__((address_space(3))) i32x4*)(l), 0, 0);
#else
  *(uint4*)(void*)l = *(const uint4*)(const void*)g;
#endif
}

#if HAVE_TDM
typedef unsigned u32x4 __attribute__((vector_size(16)));
typedef int      s32x8 __attribute__((vector_size(32)));

__device__ __forceinline__ unsigned lds_off(const void* p) {
  return (unsigned)(unsigned long long)
      (__attribute__((address_space(3))) const void*)p;
}
// 2D tile DMA: rows x tile_w bf16 elements, row stride = stride_elems,
// packed contiguously into LDS at lds_byte_off. Issued once per wave.
__device__ __forceinline__ void tdm_load_2d(const __bf16* gaddr, unsigned lds_byte_off,
                                            unsigned tile_w, unsigned tile_h,
                                            unsigned stride_elems) {
  unsigned long long ga = (unsigned long long)gaddr;
  u32x4 g0;
  g0[0] = 1u;                                            // count=1, user D#
  g0[1] = lds_byte_off;                                  // lds_addr
  g0[2] = (unsigned)(ga & 0xffffffffu);                  // global_addr[31:0]
  g0[3] = (unsigned)((ga >> 32) & 0x01ffffffu) | (2u << 30); // addr[56:32], type=2
  s32x8 g1;
  unsigned dim0 = stride_elems;                          // tensor width (no OOB)
  unsigned dim1 = 1u << 20;                              // plenty of rows
  g1[0] = 0x00010000;                                    // data_size=1 -> 2 bytes
  g1[1] = (int)((dim0 & 0xffffu) << 16);                 // tensor_dim0 lo16
  g1[2] = (int)((dim0 >> 16) | ((dim1 & 0xffffu) << 16));// dim0 hi | dim1 lo
  g1[3] = (int)((dim1 >> 16) | (tile_w << 16));          // dim1 hi | tile_dim0
  g1[4] = (int)(tile_h & 0xffffu);                       // tile_dim1, tile_dim2=0
  g1[5] = (int)stride_elems;                             // tensor_dim0_stride lo32
  g1[6] = 0;
  g1[7] = 0;
  i32x4 z4 = {0, 0, 0, 0};
  s32x8 z8 = {0, 0, 0, 0, 0, 0, 0, 0};
  __builtin_amdgcn_tensor_load_to_lds(g0, g1, z4, z4, z8, 0);   // 6-arg form
}
#define WAIT_TENSOR(n) __builtin_amdgcn_s_wait_tensorcnt(n)
#endif

// ---------------- WMMA fragment helpers (bf16 16x16x32) -------------------
__device__ __forceinline__ v16bf frag_a_lds(const __bf16* p, int ld, int m0, int k0) {
  int lane = threadIdx.x & 31, half = lane >> 4, r = lane & 15;
  v16bf a;
#pragma unroll
  for (int e = 0; e < 16; ++e) {
    int k = e + ((e & 8) ? 8 : 0) + half * 8;
    a[e] = p[(m0 + r) * ld + k0 + k];
  }
  return a;
}
__device__ __forceinline__ v16bf frag_b_lds(const __bf16* p, int ld, int k0, int n0) {
  int lane = threadIdx.x & 31, half = lane >> 4, r = lane & 15;
  v16bf b;
#pragma unroll
  for (int e = 0; e < 16; ++e) {
    int k = e + ((e & 8) ? 8 : 0) + half * 8;
    b[e] = p[(k0 + k) * ld + n0 + r];
  }
  return b;
}
__device__ __forceinline__ v8f wmma_bf16(v16bf a, v16bf b, v8f c) {
  return __builtin_amdgcn_wmma_f32_16x16x32_bf16(false, a, false, b, (short)0, c, false, false);
}

// ---------------- elementwise f32 -> bf16 ---------------------------------
__global__ void k_cvt_bf16(const float* __restrict__ in, __bf16* __restrict__ out, int n) {
  int i = blockIdx.x * blockDim.x + threadIdx.x;
  int stride = gridDim.x * blockDim.x;
  for (; i < n; i += stride) out[i] = (__bf16)in[i];
}
// f32 [N][K] -> bf16 [K][N]  (one-time weight transpose for GEMM B tiles)
__global__ void k_cvt_bf16_t(const float* __restrict__ in, __bf16* __restrict__ out,
                             int N, int K) {
  int i = blockIdx.x * blockDim.x + threadIdx.x;
  int stride = gridDim.x * blockDim.x;
  int total = N * K;
  for (; i < total; i += stride) {
    int n = i / K, k = i % K;
    out[(size_t)k * N + n] = (__bf16)in[i];
  }
}

// ---------------- generic Y = act(X @ W^T), bf16 WMMA ---------------------
// X: [M,K] bf16, Wt: [K,N] bf16 (pre-transposed W), Y: [M,N] f32. 64x64 tile,
// double-buffered LDS; staging via TDM descriptors (wave0) or async copies.
__global__ __launch_bounds__(256) void k_gemm_bf16(const __bf16* __restrict__ X,
                                                   const __bf16* __restrict__ Wt,
                                                   float* __restrict__ Y,
                                                   int Mm, int Nn, int Kk, int act) {
  __shared__ __align__(16) __bf16 As[2][64][32];
  __shared__ __align__(16) __bf16 Bs[2][32][64];
  int m0 = blockIdx.y * 64, n0 = blockIdx.x * 64;
  int tid = threadIdx.x, w = tid >> 5;
  int tm = (w & 3) * 16, tnb = (w >> 2) * 32;
  v8f acc0 = {}, acc1 = {};
  int nIt = Kk >> 5;
#if HAVE_TDM
  const __bf16* xt = X  + (size_t)m0 * Kk;         // A tile origin (row m0, k=0)
  const __bf16* wt = Wt + n0;                      // B tile origin (k=0, col n0)
  if (tid < 32) {                                  // wave 0 drives the TDM
    tdm_load_2d(xt,            lds_off(&As[0][0][0]), 32, 64, (unsigned)Kk);
    tdm_load_2d(wt,            lds_off(&Bs[0][0][0]), 64, 32, (unsigned)Nn);
  }
  for (int it = 0; it < nIt; ++it) {
    int cur = it & 1;
    if (tid < 32) {
      if (it + 1 < nIt) {
        tdm_load_2d(xt + (size_t)(it + 1) * 32,      lds_off(&As[cur ^ 1][0][0]),
                    32, 64, (unsigned)Kk);
        tdm_load_2d(wt + (size_t)(it + 1) * 32 * Nn, lds_off(&Bs[cur ^ 1][0][0]),
                    64, 32, (unsigned)Nn);
        WAIT_TENSOR(2);                            // tile `it` complete (in-order)
      } else {
        WAIT_TENSOR(0);
      }
    }
    __syncthreads();
    v16bf a  = frag_a_lds(&As[cur][0][0], 32, tm, 0);
    v16bf b0 = frag_b_lds(&Bs[cur][0][0], 64, 0, tnb);
    v16bf b1 = frag_b_lds(&Bs[cur][0][0], 64, 0, tnb + 16);
    acc0 = wmma_bf16(a, b0, acc0);
    acc1 = wmma_bf16(a, b1, acc1);
    __syncthreads();
  }
#else
  int ra = tid >> 2, ca = (tid & 3) * 8;
  int kb = tid >> 3, nb = (tid & 7) * 8;
  const __bf16* xs = X  + (size_t)(m0 + ra) * Kk + ca;
  const __bf16* wsb = Wt + (size_t)kb * Nn + n0 + nb;
  cp16(xs, &As[0][ra][ca]);
  cp16(wsb, &Bs[0][kb][nb]);
  for (int it = 0; it < nIt; ++it) {
    int cur = it & 1;
    if (it + 1 < nIt) {
      cp16(xs + (size_t)(it + 1) * 32,       &As[cur ^ 1][ra][ca]);
      cp16(wsb + (size_t)(it + 1) * 32 * Nn, &Bs[cur ^ 1][kb][nb]);
      if (it + 2 < nIt) {
        __builtin_prefetch(xs + (size_t)(it + 2) * 32, 0, 3);
        __builtin_prefetch(wsb + (size_t)(it + 2) * 32 * Nn, 0, 3);
      }
      WAIT_ASYNC(2);
    } else {
      WAIT_ASYNC(0);
    }
    __syncthreads();
    v16bf a  = frag_a_lds(&As[cur][0][0], 32, tm, 0);
    v16bf b0 = frag_b_lds(&Bs[cur][0][0], 64, 0, tnb);
    v16bf b1 = frag_b_lds(&Bs[cur][0][0], 64, 0, tnb + 16);
    acc0 = wmma_bf16(a, b0, acc0);
    acc1 = wmma_bf16(a, b1, acc1);
    __syncthreads();
  }
#endif
  int lane = tid & 31, half = lane >> 4, r = lane & 15;
#pragma unroll
  for (int j = 0; j < 8; ++j) {
    int rr = m0 + tm + j + half * 8;
    float v0 = acc0[j], v1 = acc1[j];
    if (act == 1) {                                  // SiLU
      v0 = v0 / (1.f + __expf(-v0));
      v1 = v1 / (1.f + __expf(-v1));
    }
    Y[(size_t)rr * Nn + n0 + tnb + r]      = v0;
    Y[(size_t)rr * Nn + n0 + tnb + 16 + r] = v1;
  }
}

// ---------------- cumlogsumexp over time, per (b,h,dk) channel ------------
__global__ void k_lcse(const float* __restrict__ kb, float* __restrict__ zb) {
  int bh = blockIdx.x, b = bh / H_, h = bh % H_;
  int d = threadIdx.x;                         // 128 threads = DK
  float m = -__builtin_inff(), s = 0.f;
  const float* kp = kb + (size_t)b * T_ * HID_ + h * DK_ + d;
  float*       zp = zb + (size_t)b * T_ * HID_ + h * DK_ + d;
  for (int t = 0; t < T_; ++t) {
    float val = kp[(size_t)t * HID_];
    if (val > m) { s = s * __expf(m - val) + 1.f; m = val; }
    else         { s += __expf(val - m); }
    zp[(size_t)t * HID_] = m + __logf(s);
  }
}

// ---------------- per-chunk U = Kx^T @ V and decay D ----------------------
__global__ __launch_bounds__(256) void k_chunk_kv(const float* __restrict__ kb,
                                                  const float* __restrict__ vb,
                                                  const float* __restrict__ zb,
                                                  float* __restrict__ U,
                                                  float* __restrict__ D) {
  int bc = blockIdx.x, c = bc % NC_, bh = bc / NC_, b = bh / H_, h = bh % H_;
  __shared__ __align__(16) __bf16 KsT[DK_][CHK_];   // transposed at staging
  __shared__ __align__(16) __bf16 Vs[CHK_][DV_];
  __shared__ float ze[DK_];
  int tid = threadIdx.x;
  size_t rowe = (size_t)(b * T_ + c * CHK_ + CHK_ - 1) * HID_ + h * DK_;
  if (tid < DK_) ze[tid] = zb[rowe + tid];
  __syncthreads();
  size_t base = (size_t)(b * T_ + c * CHK_) * HID_ + h * DK_;
  for (int idx = tid; idx < CHK_ * DK_; idx += 256) {
    int s = idx >> 7, d = idx & 127;
    KsT[d][s] = (__bf16)__expf(kb[base + (size_t)s * HID_ + d] - ze[d]);
    Vs[s][d]  = (__bf16)vb[base + (size_t)s * HID_ + d];
  }
  __syncthreads();
  int w = tid >> 5, m0 = w * 16;               // each wave: 16 rows of DK
  float* Uo = U + ((size_t)bc << 14);          // 128*128
  v8f acc[8] = {};
  for (int kk = 0; kk < CHK_; kk += 32) {
    v16bf a = frag_a_lds(&KsT[0][0], CHK_, m0, kk);   // A = Ks^T (contig rows)
#pragma unroll
    for (int i = 0; i < 8; ++i) {
      v16bf bf = frag_b_lds(&Vs[0][0], DV_, kk, i * 16);
      acc[i] = wmma_bf16(a, bf, acc[i]);
    }
  }
  int lane = tid & 31, half = lane >> 4, r = lane & 15;
#pragma unroll
  for (int i = 0; i < 8; ++i)
#pragma unroll
    for (int j = 0; j < 8; ++j)
      Uo[(size_t)(m0 + j + half * 8) * DV_ + i * 16 + r] = acc[i][j];
  if (tid < DK_) {
    float dv = 0.f;
    if (c != 0) {
      float zp = zb[(size_t)(b * T_ + c * CHK_ - 1) * HID_ + h * DK_ + tid];
      dv = __expf(zp - ze[tid]);
    }
    D[(size_t)bc * DK_ + tid] = dv;
  }
}

// ---- sequential chunk-state prefix (in place: U -> Sprev, + bf16 copy) ---
__global__ void k_combine(float* __restrict__ U, const float* __restrict__ D,
                          __bf16* __restrict__ Sb) {
  int gid = blockIdx.x * blockDim.x + threadIdx.x;
  if (gid >= BH_ * DK_ * DV_) return;
  int bh = gid >> 14, rem = gid & 16383, d = rem >> 7;
  float S = 0.f;
  for (int c = 0; c < NC_; ++c) {
    size_t idx = ((size_t)(bh * NC_ + c) << 14) + rem;
    float u  = U[idx];
    float dc = D[(size_t)(bh * NC_ + c) * DK_ + d];
    U[idx]  = S;                      // Sprev (f32, kept for reference)
    Sb[idx] = (__bf16)S;              // Sprev (bf16, consumed by attn kernel)
    S = dc * S + u;
  }
}

// ---------------- per-chunk output: Qd@Sprev + mask(Qx@Kx^T)@V ------------
__global__ __launch_bounds__(256) void k_chunk_attn(const float* __restrict__ qb,
                                                    const float* __restrict__ kb,
                                                    const float* __restrict__ vb,
                                                    const float* __restrict__ zb,
                                                    const __bf16* __restrict__ Sb,
                                                    float* __restrict__ O) {
  int bc = blockIdx.x, c = bc % NC_, bh = bc / NC_, b = bh / H_, h = bh % H_;
  __shared__ __align__(16) __bf16 Ab[CHK_][DK_];   // Qx then Qd
  __shared__ __align__(16) __bf16 Bb[CHK_][DK_];   // V
  __shared__ __align__(16) __bf16 KxT[DK_][CHK_];  // Kx^T, reused for S slices
  __shared__ __align__(16) __bf16 Pb[CHK_][CHK_];  // masked scores
  __shared__ float ze[DK_], zpv[DK_];
  int tid = threadIdx.x, w = tid >> 5;
  if (tid < DK_) {
    ze[tid]  = zb[(size_t)(b * T_ + c * CHK_ + CHK_ - 1) * HID_ + h * DK_ + tid];
    zpv[tid] = (c == 0) ? 0.f
             : zb[(size_t)(b * T_ + c * CHK_ - 1) * HID_ + h * DK_ + tid];
  }
  __syncthreads();
  size_t base = (size_t)(b * T_ + c * CHK_) * HID_ + h * DK_;
  for (int idx = tid; idx < CHK_ * DK_; idx += 256) {
    int t = idx >> 7, d = idx & 127;
    float zt = zb[base + (size_t)t * HID_ + d];
    Ab[t][d]  = (__bf16)(qb[base + (size_t)t * HID_ + d] * __expf(ze[d] - zt));
    KxT[d][t] = (__bf16)__expf(kb[base + (size_t)t * HID_ + d] - ze[d]);
  }
  __syncthreads();
  {  // scores P = Qx @ Kx^T with causal mask; 64x64 out, K=128
    int tm = (w & 3) * 16, tn0 = (w >> 2) * 32;
    v8f p0 = {}, p1 = {};
    for (int kk = 0; kk < DK_; kk += 32) {
      v16bf a  = frag_a_lds(&Ab[0][0], DK_, tm, kk);
      v16bf b0 = frag_b_lds(&KxT[0][0], CHK_, kk, tn0);
      v16bf b1 = frag_b_lds(&KxT[0][0], CHK_, kk, tn0 + 16);
      p0 = wmma_bf16(a, b0, p0);
      p1 = wmma_bf16(a, b1, p1);
    }
    int lane = tid & 31, half = lane >> 4, r = lane & 15;
#pragma unroll
    for (int j = 0; j < 8; ++j) {
      int t = tm + j + half * 8;
      int s0 = tn0 + r, s1 = tn0 + 16 + r;
      Pb[t][s0] = (__bf16)((s0 <= t) ? p0[j] : 0.f);
      Pb[t][s1] = (__bf16)((s1 <= t) ? p1[j] : 0.f);
    }
  }
  __syncthreads();
  for (int idx = tid; idx < CHK_ * DK_; idx += 256) {   // refill: Qd, V
    int t = idx >> 7, d = idx & 127;
    float zt = zb[base + (size_t)t * HID_ + d];
    float f = (c == 0) ? 0.f : __expf(zpv[d] - zt);
    Ab[t][d] = (__bf16)(qb[base + (size_t)t * HID_ + d] * f);
    Bb[t][d] = (__bf16)vb[base + (size_t)t * HID_ + d];
  }
  __syncthreads();
  int tm = (w & 3) * 16, tnb = (w >> 2) * 64;
  v8f acc[4] = {};
  for (int kk = 0; kk < CHK_; kk += 32) {               // P @ V
    v16bf a = frag_a_lds(&Pb[0][0], CHK_, tm, kk);
#pragma unroll
    for (int i = 0; i < 4; ++i) {
      v16bf bf = frag_b_lds(&Bb[0][0], DK_, kk, tnb + i * 16);
      acc[i] = wmma_bf16(a, bf, acc[i]);
    }
  }
  // Qd @ Sprev, staging Sprev (bf16) in two 64x128 slices through KxT buffer
  const __bf16* Sp = Sb + ((size_t)bc << 14);
  __bf16* Sl = &KxT[0][0];                              // reused as [64][128]
  for (int ss = 0; ss < DK_; ss += 64) {
    __syncthreads();                                    // slice buffer free
#pragma unroll
    for (int u = 0; u < 4; ++u) {
      int ch = tid * 4 + u;                             // 16B chunks: 64*128*2/16
      int rrow = ch >> 4, ccol = (ch & 15) * 8;
      cp16(Sp + (size_t)(ss + rrow) * DV_ + ccol, Sl + rrow * DV_ + ccol);
    }
    WAIT_ASYNC(0);
    __syncthreads();
    for (int kk = 0; kk < 64; kk += 32) {
      v16bf a = frag_a_lds(&Ab[0][0], DK_, tm, ss + kk);
#pragma unroll
      for (int i = 0; i < 4; ++i) {
        v16bf bf = frag_b_lds(Sl, DV_, kk, tnb + i * 16);
        acc[i] = wmma_bf16(a, bf, acc[i]);
      }
    }
  }
  int lane = tid & 31, half = lane >> 4, r = lane & 15;
#pragma unroll
  for (int i = 0; i < 4; ++i)
#pragma unroll
    for (int j = 0; j < 8; ++j) {
      int t = tm + j + half * 8;
      O[(size_t)(b * T_ + c * CHK_ + t) * HID_ + h * DV_ + tnb + i * 16 + r] =
          acc[i][j] * SCALE_;
    }
}

// ---------------- RMSNorm * gnw * swish-gate, store bf16 ------------------
__global__ __launch_bounds__(256) void k_epilogue(const float* __restrict__ O,
                                                  const float* __restrict__ G,
                                                  const float* __restrict__ gnw,
                                                  __bf16* __restrict__ ob) {
  int row = blockIdx.x, tid = threadIdx.x;
  __shared__ float red[256];
  const float* o = O + (size_t)row * HID_;
  float vals[4], ssq = 0.f;
#pragma unroll
  for (int i = 0; i < 4; ++i) { vals[i] = o[tid + i * 256]; ssq += vals[i] * vals[i]; }
  red[tid] = ssq;
  __syncthreads();
  for (int s = 128; s > 0; s >>= 1) {
    if (tid < s) red[tid] += red[tid + s];
    __syncthreads();
  }
  float rms = rsqrtf(red[0] / (float)HID_ + EPS_);
#pragma unroll
  for (int i = 0; i < 4; ++i) {
    int col = tid + i * 256;
    float g = G[(size_t)row * HID_ + col];
    float v = vals[i] * rms * gnw[col];
    v *= g / (1.f + __expf(-g));
    ob[(size_t)row * HID_ + col] = (__bf16)v;
  }
}

// --------------------------------------------------------------------------
extern "C" void kernel_launch(void* const* d_in, const int* in_sizes, int n_in,
                              void* d_out, int out_size, void* d_ws, size_t ws_size,
                              hipStream_t stream) {
  (void)in_sizes; (void)n_in; (void)out_size; (void)ws_size;
  const float* x   = (const float*)d_in[0];
  const float* Wq  = (const float*)d_in[1];
  const float* Wk  = (const float*)d_in[2];
  const float* Wv  = (const float*)d_in[3];
  const float* Wg1 = (const float*)d_in[4];
  const float* Wg2 = (const float*)d_in[5];
  const float* gnw = (const float*)d_in[6];
  const float* Wo  = (const float*)d_in[7];
  float* out = (float*)d_out;

  char* ws = (char*)d_ws;
  size_t off = 0;
  auto alloc = [&](size_t bytes) -> void* {
    off = (off + 255) & ~(size_t)255;
    void* p = ws + off;
    off += bytes;
    return p;
  };
  __bf16* xb   = (__bf16*)alloc((size_t)M_ * HID_ * 2);
  __bf16* Wqt  = (__bf16*)alloc((size_t)HID_ * HID_ * 2);
  __bf16* Wkt  = (__bf16*)alloc((size_t)HID_ * HID_ * 2);
  __bf16* Wvt  = (__bf16*)alloc((size_t)HID_ * HID_ * 2);
  __bf16* Wot  = (__bf16*)alloc((size_t)HID_ * HID_ * 2);
  __bf16* Wg1t = (__bf16*)alloc((size_t)LR_ * HID_ * 2);
  __bf16* Wg2t = (__bf16*)alloc((size_t)HID_ * LR_ * 2);
  float*  qf   = (float*)alloc((size_t)M_ * HID_ * 4);
  float*  kf   = (float*)alloc((size_t)M_ * HID_ * 4);
  float*  vf   = (float*)alloc((size_t)M_ * HID_ * 4);
  float*  zf   = (float*)alloc((size_t)M_ * HID_ * 4);
  float*  U    = (float*)alloc((size_t)BH_ * NC_ * DK_ * DV_ * 4);
  __bf16* Sb   = (__bf16*)alloc((size_t)BH_ * NC_ * DK_ * DV_ * 2);
  float*  D    = (float*)alloc((size_t)BH_ * NC_ * DK_ * 4);
  float*  G1   = (float*)alloc((size_t)M_ * LR_ * 4);
  __bf16* G1b  = (__bf16*)alloc((size_t)M_ * LR_ * 2);
  float*  G    = (float*)alloc((size_t)M_ * HID_ * 4);
  float*  O    = (float*)alloc((size_t)M_ * HID_ * 4);
  __bf16* ob   = (__bf16*)alloc((size_t)M_ * HID_ * 2);

  // conversions: activations plain, weights transposed to [K][N]
  k_cvt_bf16<<<2048, 256, 0, stream>>>(x, xb, M_ * HID_);
  k_cvt_bf16_t<<<1024, 256, 0, stream>>>(Wq,  Wqt,  HID_, HID_);
  k_cvt_bf16_t<<<1024, 256, 0, stream>>>(Wk,  Wkt,  HID_, HID_);
  k_cvt_bf16_t<<<1024, 256, 0, stream>>>(Wv,  Wvt,  HID_, HID_);
  k_cvt_bf16_t<<<1024, 256, 0, stream>>>(Wo,  Wot,  HID_, HID_);
  k_cvt_bf16_t<<<256,  256, 0, stream>>>(Wg1, Wg1t, LR_,  HID_);
  k_cvt_bf16_t<<<256,  256, 0, stream>>>(Wg2, Wg2t, HID_, LR_);

  // projections
  dim3 gproj(HID_ / 64, M_ / 64);
  k_gemm_bf16<<<gproj, 256, 0, stream>>>(xb, Wqt, qf, M_, HID_, HID_, 1); // silu
  k_gemm_bf16<<<gproj, 256, 0, stream>>>(xb, Wkt, kf, M_, HID_, HID_, 0);
  k_gemm_bf16<<<gproj, 256, 0, stream>>>(xb, Wvt, vf, M_, HID_, HID_, 0);

  // cumulative logsumexp over time
  k_lcse<<<BH_, DK_, 0, stream>>>(kf, zf);

  // chunked linear-attention
  k_chunk_kv<<<BH_ * NC_, 256, 0, stream>>>(kf, vf, zf, U, D);
  k_combine<<<(BH_ * DK_ * DV_) / 256, 256, 0, stream>>>(U, D, Sb);
  k_chunk_attn<<<BH_ * NC_, 256, 0, stream>>>(qf, kf, vf, zf, Sb, O);

  // low-rank gate
  k_gemm_bf16<<<dim3(LR_ / 64, M_ / 64), 256, 0, stream>>>(xb, Wg1t, G1, M_, LR_, HID_, 0);
  k_cvt_bf16<<<512, 256, 0, stream>>>(G1, G1b, M_ * LR_);
  k_gemm_bf16<<<dim3(HID_ / 64, M_ / 64), 256, 0, stream>>>(G1b, Wg2t, G, M_, HID_, LR_, 0);

  // rmsnorm * swish(gate), then output projection
  k_epilogue<<<M_, 256, 0, stream>>>(O, G, gnw, ob);
  k_gemm_bf16<<<dim3(HID_ / 64, M_ / 64), 256, 0, stream>>>(ob, Wot, out, M_, HID_, HID_, 0);
}
